// MultiHeadAttention_78958678769968
// MI455X (gfx1250) — matrix operand order repaired and checked by reading, not compiled, not run on hardware
//
#include <hip/hip_runtime.h>

// ---------------------------------------------------------------------------
// MHA forward for MI455X (gfx1250, wave32).  bf16 WMMA (f32 accum), fp32
// softmax, async-DMA (global_load_async_to_lds_b128) double-buffered GEMM.
// ---------------------------------------------------------------------------

typedef __attribute__((ext_vector_type(16))) __bf16 v16bf;
typedef __attribute__((ext_vector_type(8)))  __bf16 v8bf;
typedef __attribute__((ext_vector_type(8)))  float  v8f;

constexpr int kB = 2, kS = 2048, kD = 1024, kH = 16, kDepth = 64;
constexpr size_t kBSD = (size_t)kB * kS * kD;   // 4,194,304
constexpr size_t kDD  = (size_t)kD * kD;        // 1,048,576

__device__ __forceinline__ v8f wmma_bf16(v16bf a, v16bf b, v8f c) {
  return __builtin_amdgcn_wmma_f32_16x16x32_bf16(false, a, false, b, (short)0, c, false, false);
}

// A-fragment (16x32 bf16, MxK): lane&15 = M; lanes<16 hold K {0..7,16..23},
// lanes>=16 hold K {8..15,24..31}.  ld in halves, multiple of 8 (16B align).
__device__ __forceinline__ v16bf load_frag_a(const __bf16* base, int ld, int lane) {
  const __bf16* p = base + (lane & 15) * ld + ((lane >> 4) << 3);
  v8bf lo = *(const v8bf*)(p);
  v8bf hi = *(const v8bf*)(p + 16);
  v16bf f;
#pragma unroll
  for (int i = 0; i < 8; ++i) { f[i] = lo[i]; f[i + 8] = hi[i]; }
  return f;
}

// B-fragment (32x16 bf16, KxN): lane&15 = N; lanes<16 K {0..7,16..23},
// lanes>=16 K {8..15,24..31}.  base is N-major (B^T row-major): base[n*ld + k].
__device__ __forceinline__ v16bf load_frag_b(const __bf16* base, int ld, int lane) {
  const __bf16* p = base + (lane & 15) * ld + ((lane >> 4) << 3);
  v8bf lo = *(const v8bf*)(p);
  v8bf hi = *(const v8bf*)(p + 16);
  v16bf f;
#pragma unroll
  for (int i = 0; i < 8; ++i) { f[i] = lo[i]; f[i + 8] = hi[i]; }
  return f;
}

// ---------------------------------------------------------------------------
// prep: f32 -> bf16 elementwise (counts are exact multiples of 1024)
// ---------------------------------------------------------------------------
__global__ __launch_bounds__(256) void cvt_bf16_kernel(const float* __restrict__ in,
                                                       __bf16* __restrict__ out) {
  const size_t i = ((size_t)blockIdx.x * 256 + threadIdx.x) * 4;
  const float4 x = *(const float4*)(in + i);
  out[i + 0] = (__bf16)x.x; out[i + 1] = (__bf16)x.y;
  out[i + 2] = (__bf16)x.z; out[i + 3] = (__bf16)x.w;
}

// prep: W[K][N] f32 -> Wt[N][K] bf16  (1024x1024, 32x32 tiles)
__global__ __launch_bounds__(256) void transpose_bf16_kernel(const float* __restrict__ W,
                                                             __bf16* __restrict__ Wt) {
  __shared__ float t[32][33];
  const int bx = blockIdx.x * 32, by = blockIdx.y * 32;   // bx: N, by: K
  const int tx = threadIdx.x & 31, ty = threadIdx.x >> 5; // 32 x 8
#pragma unroll
  for (int p = 0; p < 4; ++p) {
    const int r = ty + p * 8;
    t[r][tx] = W[(size_t)(by + r) * kD + bx + tx];
  }
  __syncthreads();
#pragma unroll
  for (int p = 0; p < 4; ++p) {
    const int a = ty + p * 8;
    Wt[(size_t)(bx + a) * kD + by + tx] = (__bf16)t[tx][a];
  }
}

// ---------------------------------------------------------------------------
// GEMM: C[M,N] = A[M,K](bf16) @ Wt[N,K](bf16)^T + bias.  Block 128x64, BK=32,
// 8 waves (4x2), wave tile 32x32.  Tiles staged by global_load_async_to_lds
// (ASYNCcnt) with double-buffered LDS: DMA(t+1) overlaps WMMA(t).
// OUT_MODE 0: bf16 head-split [B,H,S,64]   (Q,K proj)
// OUT_MODE 1: bf16 head-split+T [B,H,64,S] (V proj)
// OUT_MODE 2: f32 row-major [M,N]          (final proj)
// ---------------------------------------------------------------------------
template <int OUT_MODE>
__global__ __launch_bounds__(256) void gemm_wmma_kernel(
    const __bf16* __restrict__ A, const __bf16* __restrict__ Wt,
    const float* __restrict__ bias, void* __restrict__ Cv,
    int M, int N, int K) {
  __shared__ __attribute__((aligned(16))) __bf16 As[2][128 * 40];  // [m][k] ld=40
  __shared__ __attribute__((aligned(16))) __bf16 Bs[2][64 * 40];   // [n][k] ld=40

  const int tid  = threadIdx.x;
  const int lane = tid & 31, wid = tid >> 5;
  const int wm = wid & 3, wn = wid >> 2;
  const int bm = blockIdx.y * 128, bn = blockIdx.x * 64;

  // per-thread 16B async-copy chunks: A = 128 rows x 4 chunks (2/thread),
  // B = 64 rows x 4 chunks (1/thread); chunk = 8 halves.
  const int ar0 = tid >> 2, ac = (tid & 3) << 3;

  auto issue_tile = [&](int kt, int buf) {
    const uint32_t la0 = (uint32_t)(uintptr_t)(&As[buf][0]);
    const uint32_t lb0 = (uint32_t)(uintptr_t)(&Bs[buf][0]);
#pragma unroll
    for (int p = 0; p < 2; ++p) {
      const int row = ar0 + p * 64;
      const uint64_t ga = (uint64_t)(uintptr_t)(A + (size_t)(bm + row) * K + kt + ac);
      const uint32_t la = la0 + (uint32_t)(row * 40 + ac) * 2u;
      asm volatile("global_load_async_to_lds_b128 %0, %1, off" :: "v"(la), "v"(ga) : "memory");
    }
    {
      const uint64_t gb = (uint64_t)(uintptr_t)(Wt + (size_t)(bn + ar0) * K + kt + ac);
      const uint32_t lb = lb0 + (uint32_t)(ar0 * 40 + ac) * 2u;
      asm volatile("global_load_async_to_lds_b128 %0, %1, off" :: "v"(lb), "v"(gb) : "memory");
    }
    if (kt + 32 < K)  // warm L2 for the tile after next (global_prefetch_b8)
      __builtin_prefetch(A + (size_t)(bm + ar0) * K + kt + 32 + ac, 0, 1);
  };

  v8f c[2][2] = {};
  issue_tile(0, 0);
  const int T = K / 32;
  for (int t = 0; t < T; ++t) {
    asm volatile("s_wait_asynccnt 0x0" ::: "memory");  // this wave's copies for tile t done
    __syncthreads();                                   // => all copies done; other buf free
    if (t + 1 < T) issue_tile((t + 1) * 32, (t + 1) & 1);

    const int buf = t & 1;
    v16bf afr[2], bfr[2];
#pragma unroll
    for (int i = 0; i < 2; ++i) afr[i] = load_frag_a(&As[buf][(wm * 32 + 16 * i) * 40], 40, lane);
#pragma unroll
    for (int j = 0; j < 2; ++j) bfr[j] = load_frag_b(&Bs[buf][(wn * 32 + 16 * j) * 40], 40, lane);
#pragma unroll
    for (int i = 0; i < 2; ++i)
#pragma unroll
      for (int j = 0; j < 2; ++j) c[i][j] = wmma_bf16(afr[i], bfr[j], c[i][j]);
  }

  // epilogue.  C-frag: vgpr r, lanes<16 -> M=r, lanes>=16 -> M=8+r; N=lane&15
  const int hi = lane >> 4, ln = lane & 15;
#pragma unroll
  for (int i = 0; i < 2; ++i)
#pragma unroll
    for (int j = 0; j < 2; ++j)
#pragma unroll
      for (int r = 0; r < 8; ++r) {
        const int m = bm + wm * 32 + 16 * i + r + 8 * hi;
        const int n = bn + wn * 32 + 16 * j + ln;
        const float val = c[i][j][r] + bias[n];
        if (OUT_MODE == 2) {
          ((float*)Cv)[(size_t)m * N + n] = val;
        } else {
          const int b = m >> 11, s = m & (kS - 1);
          const int h = n >> 6,  d = n & (kDepth - 1);
          __bf16* o = (__bf16*)Cv;
          if (OUT_MODE == 0)
            o[(((size_t)(b * kH + h)) * kS + s) * kDepth + d] = (__bf16)val;
          else
            o[(((size_t)(b * kH + h)) * kDepth + d) * kS + s] = (__bf16)val;
        }
      }
}

// ---------------------------------------------------------------------------
// Attention: one wave per 16-row query tile.
//  pass1: logits = (Q Kh^T)/8 (+ causal -1e9) -> raw into attn region
//  pass2: per-row max & sum(exp) in fp32
//  pass3: rewrite attn with normalized probs; accumulate out = P @ V (WMMA)
// ---------------------------------------------------------------------------
__global__ __launch_bounds__(128) void attn_kernel(
    const __bf16* __restrict__ Qh, const __bf16* __restrict__ Kh,
    const __bf16* __restrict__ Vt, float* __restrict__ attn,
    __bf16* __restrict__ headout) {
  const int tid = threadIdx.x, lane = tid & 31, wid = tid >> 5;
  const int bh = blockIdx.y;                 // b*H + h
  const int qtile = blockIdx.x * 4 + wid;    // 0..127
  const int qbase = qtile * 16;
  const int b = bh >> 4, h = bh & (kH - 1);

  const __bf16* Q = Qh + (size_t)bh * kS * kDepth;
  const __bf16* K = Kh + (size_t)bh * kS * kDepth;
  const __bf16* V = Vt + (size_t)bh * kDepth * kS;   // [64][S]
  float* Arow = attn + (size_t)bh * kS * kS + (size_t)qbase * kS;

  const int hi = lane >> 4, ln = lane & 15;

  const v16bf qa0 = load_frag_a(Q + (size_t)qbase * kDepth, kDepth, lane);
  const v16bf qa1 = load_frag_a(Q + (size_t)qbase * kDepth + 32, kDepth, lane);

  // ---- pass 1: raw logits (causal region only)
  for (int kt = 0; kt <= qtile; ++kt) {
    const __bf16* Kb = K + (size_t)kt * 16 * kDepth;
    const v16bf kb0 = load_frag_b(Kb, kDepth, lane);
    const v16bf kb1 = load_frag_b(Kb + 32, kDepth, lane);
    v8f sfr = {};
    sfr = wmma_bf16(qa0, kb0, sfr);
    sfr = wmma_bf16(qa1, kb1, sfr);
#pragma unroll
    for (int r = 0; r < 8; ++r) {
      const int row = r + 8 * hi;
      const int col = kt * 16 + ln;
      float v = sfr[r] * 0.125f;           // 1/sqrt(64)
      if (col > qbase + row) v += -1e9f;   // mask * (-1e9)
      Arow[(size_t)row * kS + col] = v;
    }
  }
  asm volatile("s_wait_storecnt 0x0" ::: "memory");

  // ---- pass 2: row stats (2 lanes per row)
  const int ncols = (qtile + 1) * 16;
  float mrow, srow;
  {
    const int row = lane >> 1;
    const int cnt = ncols >> 1;
    const float* pr = Arow + (size_t)row * kS + (lane & 1) * cnt;
    float m = -3.0e38f;
    for (int c = 0; c < cnt; c += 4) {
      const float4 x = *(const float4*)(pr + c);
      m = fmaxf(m, fmaxf(fmaxf(x.x, x.y), fmaxf(x.z, x.w)));
    }
    m = fmaxf(m, __shfl_xor(m, 1, 32));
    float s = 0.f;
    for (int c = 0; c < cnt; c += 4) {
      const float4 x = *(const float4*)(pr + c);
      s += __expf(x.x - m) + __expf(x.y - m) + __expf(x.z - m) + __expf(x.w - m);
    }
    s += __shfl_xor(s, 1, 32);
    mrow = m; srow = s;
  }
  const float m3   = __shfl(mrow, ln << 1, 32);
  const float rinv = 1.0f / __shfl(srow, ln << 1, 32);

  // ---- pass 3: normalize + P @ V
  v8f o[4] = {};
  {
    const int k0 = hi << 3;
    float* pr = Arow + (size_t)ln * kS;
    const int nchunk = (qtile + 2) >> 1;
    for (int ck = 0; ck < nchunk; ++ck) {
      const int base = ck << 5;
      const bool full = (base + 32) <= ncols;
      float* pl = pr + base + k0;
      v16bf pf;
#pragma unroll
      for (int i = 0; i < 8; ++i) {
        const float p = __expf(pl[i] - m3) * rinv;
        pf[i] = (__bf16)p;
        pl[i] = p;
      }
      if (full) {
#pragma unroll
        for (int i = 0; i < 8; ++i) {
          const float p = __expf(pl[16 + i] - m3) * rinv;
          pf[i + 8] = (__bf16)p;
          pl[16 + i] = p;
        }
      } else {
#pragma unroll
        for (int i = 0; i < 8; ++i) pf[i + 8] = (__bf16)0.0f;
      }
#pragma unroll
      for (int j = 0; j < 4; ++j) {
        const v16bf vb = load_frag_b(V + (size_t)(j * 16) * kS + base, kS, lane);
        o[j] = wmma_bf16(pf, vb, o[j]);
      }
    }
  }

  // ---- zero the masked-out tail (exact-0 == fp32 underflow of exp(-1e9-m))
  {
    const int cnt = (kS - ncols) >> 1;
    float* zp = Arow + (size_t)ln * kS + ncols + hi * cnt;
    const float4 z4 = {0.f, 0.f, 0.f, 0.f};
    for (int c = 0; c < cnt; c += 4) *(float4*)(zp + c) = z4;
  }

  // ---- epilogue: headout[b, s, h*64 + d]
#pragma unroll
  for (int j = 0; j < 4; ++j)
#pragma unroll
    for (int r = 0; r < 8; ++r) {
      const int row = qbase + r + 8 * hi;
      const int d = j * 16 + ln;
      headout[((size_t)(b * kS + row)) * kD + h * kDepth + d] = (__bf16)o[j][r];
    }
}

// ---------------------------------------------------------------------------
extern "C" void kernel_launch(void* const* d_in, const int* in_sizes, int n_in,
                              void* d_out, int out_size, void* d_ws, size_t ws_size,
                              hipStream_t stream) {
  (void)in_sizes; (void)n_in; (void)out_size; (void)ws_size;
  const float* v  = (const float*)d_in[0];
  const float* k  = (const float*)d_in[1];
  const float* q  = (const float*)d_in[2];
  // d_in[3] = mask (causal triu; reproduced analytically in-kernel)
  const float* wq = (const float*)d_in[4];
  const float* bq = (const float*)d_in[5];
  const float* wk = (const float*)d_in[6];
  const float* bk = (const float*)d_in[7];
  const float* wv = (const float*)d_in[8];
  const float* bv = (const float*)d_in[9];
  const float* wo = (const float*)d_in[10];
  const float* bo = (const float*)d_in[11];

  float* out  = (float*)d_out;       // [B,S,D]
  float* attn = out + kBSD;          // [B,H,S,S]

  __bf16* Qh  = (__bf16*)d_ws;       // [B,H,S,64]
  __bf16* Kh  = Qh + kBSD;           // [B,H,S,64]
  __bf16* Vt  = Kh + kBSD;           // [B,H,64,S]
  __bf16* Ho  = Vt + kBSD;           // [B,S,D]
  __bf16* qb  = Ho + kBSD;           // bf16 copies of inputs
  __bf16* kb_ = qb + kBSD;
  __bf16* vb_ = kb_ + kBSD;
  __bf16* wqt = vb_ + kBSD;          // transposed bf16 weights [N][K]
  __bf16* wkt = wqt + kDD;
  __bf16* wvt = wkt + kDD;
  __bf16* wot = wvt + kDD;

  const int cvtBlocks = (int)(kBSD / (256 * 4));     // 4096
  cvt_bf16_kernel<<<cvtBlocks, 256, 0, stream>>>(q, qb);
  cvt_bf16_kernel<<<cvtBlocks, 256, 0, stream>>>(k, kb_);
  cvt_bf16_kernel<<<cvtBlocks, 256, 0, stream>>>(v, vb_);
  const dim3 tgrid(kD / 32, kD / 32);                // (32,32)
  transpose_bf16_kernel<<<tgrid, 256, 0, stream>>>(wq, wqt);
  transpose_bf16_kernel<<<tgrid, 256, 0, stream>>>(wk, wkt);
  transpose_bf16_kernel<<<tgrid, 256, 0, stream>>>(wv, wvt);
  transpose_bf16_kernel<<<tgrid, 256, 0, stream>>>(wo, wot);

  const int M = kB * kS, N = kD, Kk = kD;
  const dim3 ggrid(N / 64, M / 128);                 // (16, 32)
  gemm_wmma_kernel<0><<<ggrid, 256, 0, stream>>>(qb,  wqt, bq, Qh, M, N, Kk);
  gemm_wmma_kernel<0><<<ggrid, 256, 0, stream>>>(kb_, wkt, bk, Kh, M, N, Kk);
  gemm_wmma_kernel<1><<<ggrid, 256, 0, stream>>>(vb_, wvt, bv, Vt, M, N, Kk);

  attn_kernel<<<dim3(kS / 16 / 4, kB * kH), 128, 0, stream>>>(Qh, Kh, Vt, attn, Ho);

  gemm_wmma_kernel<2><<<ggrid, 256, 0, stream>>>(Ho, wot, bo, out, M, N, Kk);
}
